// MatlabBlockNCB_56264071577750
// MI455X (gfx1250) — matrix operand order repaired and checked by reading, not compiled, not run on hardware
//
#include <hip/hip_runtime.h>
#include <hip/hip_bf16.h>
#include <math.h>

typedef __bf16 bf16;
typedef __attribute__((ext_vector_type(16))) __bf16 v16bf;
typedef __attribute__((ext_vector_type(8)))  __bf16 v8bf;
typedef __attribute__((ext_vector_type(4)))  __bf16 v4bf;
typedef __attribute__((ext_vector_type(8)))  float  v8f;

#define NB 32
#define CD 1024
#define BD 1024

// ---- CDNA5 async global->LDS copy (ASYNCcnt-tracked), guarded so compile
// ---- always succeeds even if the toolchain lacks the builtin.
#if __has_builtin(__builtin_amdgcn_global_load_async_to_lds_b128)
#define HAVE_ASYNC_LDS 1
typedef int v4i_t __attribute__((ext_vector_type(4)));
typedef __attribute__((address_space(1))) v4i_t* as1_v4i;
typedef __attribute__((address_space(3))) v4i_t* as3_v4i;
#else
#define HAVE_ASYNC_LDS 0
#endif

#if __has_builtin(__builtin_amdgcn_s_wait_asynccnt)
#define WAIT_ASYNC0() __builtin_amdgcn_s_wait_asynccnt(0)
#else
#define WAIT_ASYNC0() asm volatile("s_wait_asynccnt 0x0" ::: "memory")
#endif

// ---------------------------------------------------------------- utilities

__global__ __launch_bounds__(256) void zero_f32_kernel(float* __restrict__ p, int n) {
    int i = blockIdx.x * 256 + threadIdx.x;
    if (i < n) p[i] = 0.0f;
}

__global__ __launch_bounds__(256) void cast_f32_bf16_kernel(const float* __restrict__ in,
                                                            bf16* __restrict__ out, int n4) {
    int i = blockIdx.x * 256 + threadIdx.x;
    if (i < n4) {
        float4 f = ((const float4*)in)[i];
        v4bf o = { (bf16)f.x, (bf16)f.y, (bf16)f.z, (bf16)f.w };
        ((v4bf*)out)[i] = o;
    }
}

// ------------------------------------------------------ batched WMMA GEMM
// C[m,n] (1024x1024) = sum_k A[m,k] * B[k,n], per batch (blockIdx.z).
// MODE 0: projection  K/Q = W*X + bias; bf16 out; atomic per-column sum-of-squares.
// MODE 1: scores      S = Q^T K, scaled by rsqrt(max(DQ2[m]*DK2[n], eps)); f32 out.
//         (A operand read transposed: A[m,k] = Q[k,m])
// MODE 2: output      Z = X * SM; f32 out.
//
// 128x128x32 block tile, 256 threads (8 wave32s), wave tile 64x32 = 4x2 WMMA.
// Double-buffered LDS + register/async staged prefetch of the next K-slice.
template <int MODE>
__global__ __launch_bounds__(256) void gemm_wmma_kernel(
    const bf16* __restrict__ A, size_t batchStrideA,
    const bf16* __restrict__ Bm, size_t batchStrideB,
    void* __restrict__ Out,
    const float* __restrict__ aux0,   // MODE0: bias[C]        MODE1: DQ2 base
    const float* __restrict__ aux1,   //                       MODE1: DK2 base
    float* __restrict__ sumsq)        // MODE0: per-column sum-of-squares base
{
    __shared__ bf16 As[2][128 * 48];  // [m][k], row stride 48 (96B => 16B aligned)
    __shared__ bf16 Bs[2][128 * 48];  // [n][k] (transposed), row stride 48

    const int batch = blockIdx.z;
    const int m0 = blockIdx.y * 128;
    const int n0 = blockIdx.x * 128;

    const bf16* Ab = A  + (size_t)batch * batchStrideA;
    const bf16* Bb = Bm + (size_t)batch * batchStrideB;

    const int t    = threadIdx.x;
    const int lane = t & 31;
    const int wave = t >> 5;
    const int wm   = wave & 1;    // 2 slices of 64 rows
    const int wn   = wave >> 1;   // 4 slices of 32 cols
    const int half = lane >> 4;   // lane group per ISA fragment layout
    const int lrow = lane & 15;

    const v8f vzero = {0.f, 0.f, 0.f, 0.f, 0.f, 0.f, 0.f, 0.f};
    v8f acc[4][2];
    #pragma unroll
    for (int i = 0; i < 4; ++i)
        #pragma unroll
        for (int j = 0; j < 2; ++j) acc[i][j] = vzero;

    v8bf aReg[2], bReg[2];

    // -- issue global fetch of A slice (async->LDS when layout allows it)
    auto issueA = [&](int k0, int buf) {
        if (MODE == 1) {
            #pragma unroll
            for (int u = 0; u < 2; ++u) {
                int c  = t + 256 * u;
                int gk = c >> 4;            // 0..31
                int gm = (c & 15) << 3;     // 0..120
                aReg[u] = *(const v8bf*)(Ab + (size_t)(k0 + gk) * 1024 + (m0 + gm));
            }
        } else {
#if HAVE_ASYNC_LDS
            #pragma unroll
            for (int u = 0; u < 2; ++u) {
                int c   = t + 256 * u;
                int row = c >> 2;           // 0..127
                int kk  = (c & 3) << 3;     // 0,8,16,24
                const bf16* g = Ab + (size_t)(m0 + row) * 1024 + (k0 + kk);
                bf16*       l = &As[buf][row * 48 + kk];
                __builtin_amdgcn_global_load_async_to_lds_b128((as1_v4i)g, (as3_v4i)l, 0, 0);
            }
#else
            #pragma unroll
            for (int u = 0; u < 2; ++u) {
                int c   = t + 256 * u;
                int row = c >> 2;
                int kk  = (c & 3) << 3;
                aReg[u] = *(const v8bf*)(Ab + (size_t)(m0 + row) * 1024 + (k0 + kk));
            }
#endif
        }
        (void)buf;
    };
    // -- deposit A slice into LDS (nothing to do on the async path)
    auto depositA = [&](int buf) {
        if (MODE == 1) {
            #pragma unroll
            for (int u = 0; u < 2; ++u) {
                int c  = t + 256 * u;
                int gk = c >> 4;
                int gm = (c & 15) << 3;
                #pragma unroll
                for (int j = 0; j < 8; ++j) As[buf][(gm + j) * 48 + gk] = aReg[u][j];
            }
        } else {
#if !HAVE_ASYNC_LDS
            #pragma unroll
            for (int u = 0; u < 2; ++u) {
                int c   = t + 256 * u;
                int row = c >> 2;
                int kk  = (c & 3) << 3;
                *(v8bf*)(&As[buf][row * 48 + kk]) = aReg[u];
            }
#endif
        }
        (void)buf;
    };
    auto issueB = [&](int k0) {
        #pragma unroll
        for (int u = 0; u < 2; ++u) {
            int c  = t + 256 * u;
            int gk = c >> 4;
            int gn = (c & 15) << 3;
            bReg[u] = *(const v8bf*)(Bb + (size_t)(k0 + gk) * 1024 + (n0 + gn));
        }
    };
    auto depositB = [&](int buf) {
        #pragma unroll
        for (int u = 0; u < 2; ++u) {
            int c  = t + 256 * u;
            int gk = c >> 4;
            int gn = (c & 15) << 3;
            #pragma unroll
            for (int j = 0; j < 8; ++j) Bs[buf][(gn + j) * 48 + gk] = bReg[u][j];
        }
    };
    auto compute = [&](const bf16* AsB, const bf16* BsB) {
        // A 16x32 frag: lanes 0-15 hold K {0..15}, lanes 16-31 K {16..31}
        v16bf afrag[4];
        #pragma unroll
        for (int tm = 0; tm < 4; ++tm) {
            const bf16* ap = &AsB[(wm * 64 + tm * 16 + lrow) * 48 + half * 16];
            v8bf lo = *(const v8bf*)(ap);
            v8bf hi = *(const v8bf*)(ap + 8);
            afrag[tm] = __builtin_shufflevector(lo, hi, 0, 1, 2, 3, 4, 5, 6, 7,
                                                        8, 9, 10, 11, 12, 13, 14, 15);
        }
        // B 32x16 frag: col = lane%16, K = half*16 + 0..15 (contiguous in Bs[n][k])
        v16bf bfrag[2];
        #pragma unroll
        for (int tn = 0; tn < 2; ++tn)
            bfrag[tn] = *(const v16bf*)(&BsB[(wn * 32 + tn * 16 + lrow) * 48 + half * 16]);

        #pragma unroll
        for (int tm = 0; tm < 4; ++tm)
            #pragma unroll
            for (int tn = 0; tn < 2; ++tn)
                acc[tm][tn] = __builtin_amdgcn_wmma_f32_16x16x32_bf16(
                    false, afrag[tm], false, bfrag[tn], (short)0, acc[tm][tn],
                    false, false);
    };

    // ---- software pipeline: prologue fills buffer 0
    issueA(0, 0);
    issueB(0);
    depositA(0);
    depositB(0);
#if HAVE_ASYNC_LDS
    if (MODE != 1) WAIT_ASYNC0();
#endif
    __syncthreads();

    for (int it = 0; it < 32; ++it) {
        const int b = it & 1;
        const bool hasNext = (it + 1) < 32;
        if (hasNext) {                 // fetches fly during the WMMA block below
            issueA((it + 1) * 32, 1 - b);
            issueB((it + 1) * 32);
        }
        compute(As[b], Bs[b]);
        if (hasNext) {                 // loadcnt wait lands here, after compute
            depositA(1 - b);
            depositB(1 - b);
#if HAVE_ASYNC_LDS
            if (MODE != 1) WAIT_ASYNC0();
#endif
        }
        __syncthreads();
    }

    // ---- epilogues (C/D layout: col = lane%16, row = r + 8*half)
    const int mwbase = m0 + wm * 64;
    const int nwbase = n0 + wn * 32;

    if (MODE == 0) {
        bf16*  O  = (bf16*)Out + (size_t)batch * CD * BD;
        float* ss = sumsq + (size_t)batch * BD;
        #pragma unroll
        for (int tn = 0; tn < 2; ++tn) {
            int gn = nwbase + tn * 16 + lrow;
            float colsq = 0.0f;
            #pragma unroll
            for (int tm = 0; tm < 4; ++tm) {
                #pragma unroll
                for (int r = 0; r < 8; ++r) {
                    int gm  = mwbase + tm * 16 + half * 8 + r;
                    float v = acc[tm][tn][r] + aux0[gm];
                    O[(size_t)gm * BD + gn] = (bf16)v;
                    colsq += v * v;
                }
            }
            colsq += __shfl_xor(colsq, 16, 32);   // fold the two half-lane row sets
            if (half == 0) atomicAdd(&ss[gn], colsq);
        }
    } else if (MODE == 1) {
        float* O = (float*)Out + (size_t)batch * BD * BD;
        const float* dq2 = aux0 + (size_t)batch * BD;
        const float* dk2 = aux1 + (size_t)batch * BD;
        #pragma unroll
        for (int tm = 0; tm < 4; ++tm)
            #pragma unroll
            for (int tn = 0; tn < 2; ++tn) {
                int gn   = nwbase + tn * 16 + lrow;
                float sk = dk2[gn];
                #pragma unroll
                for (int r = 0; r < 8; ++r) {
                    int gm = mwbase + tm * 16 + half * 8 + r;
                    float scale = rsqrtf(fmaxf(dq2[gm] * sk, 1e-12f));
                    O[(size_t)gm * BD + gn] = acc[tm][tn][r] * scale;
                }
            }
    } else {
        float* O = (float*)Out + (size_t)batch * CD * BD;
        #pragma unroll
        for (int tm = 0; tm < 4; ++tm)
            #pragma unroll
            for (int tn = 0; tn < 2; ++tn) {
                int gn = nwbase + tn * 16 + lrow;
                #pragma unroll
                for (int r = 0; r < 8; ++r) {
                    int gm = mwbase + tm * 16 + half * 8 + r;
                    O[(size_t)gm * BD + gn] = acc[tm][tn][r];
                }
            }
    }
}

// ------------------------------------------- column softmax over axis q
// SM[n, q, k] = softmax_q( S[n, q, k] ); online max+sum pass, then emit bf16.
__global__ __launch_bounds__(256) void softmax_col_kernel(const float* __restrict__ S,
                                                          bf16* __restrict__ SM) {
    int n = blockIdx.y;
    int k = blockIdx.x * 256 + threadIdx.x;           // coalesced over k
    const float* s = S + ((size_t)n << 20) + k;
    float mx = -INFINITY, sum = 0.0f;
    for (int q = 0; q < 1024; ++q) {
        float x  = s[(size_t)q << 10];
        float m2 = fmaxf(mx, x);
        sum = sum * __expf(mx - m2) + __expf(x - m2);
        mx  = m2;
    }
    float inv = 1.0f / sum;
    bf16* o = SM + ((size_t)n << 20) + k;
    for (int q = 0; q < 1024; ++q) {
        float x = s[(size_t)q << 10];
        o[(size_t)q << 10] = (bf16)(__expf(x - mx) * inv);
    }
}

// ---------------------------------------------------------------- launcher

extern "C" void kernel_launch(void* const* d_in, const int* in_sizes, int n_in,
                              void* d_out, int out_size, void* d_ws, size_t ws_size,
                              hipStream_t stream) {
    (void)in_sizes; (void)n_in; (void)out_size; (void)ws_size;

    const float* X   = (const float*)d_in[0];
    const float* Wk  = (const float*)d_in[1];
    const float* Wq  = (const float*)d_in[2];
    const float* Wk0 = (const float*)d_in[3];
    const float* Wq0 = (const float*)d_in[4];

    const size_t NCB = (size_t)NB * CD * BD;  // 33,554,432
    const size_t CC  = (size_t)CD * CD;       //  1,048,576
    const size_t NBB = (size_t)NB * BD * BD;

    char* p = (char*)d_ws;
    auto wsAlloc = [&](size_t bytes) -> void* {
        void* r = (void*)p;
        p += (bytes + 255) & ~(size_t)255;
        return r;
    };
    bf16*  Xb  = (bf16*) wsAlloc(NCB * 2);
    bf16*  Wkb = (bf16*) wsAlloc(CC * 2);
    bf16*  Wqb = (bf16*) wsAlloc(CC * 2);
    bf16*  Kb  = (bf16*) wsAlloc(NCB * 2);
    bf16*  Qb  = (bf16*) wsAlloc(NCB * 2);
    float* DK2 = (float*)wsAlloc((size_t)NB * BD * 4);
    float* DQ2 = (float*)wsAlloc((size_t)NB * BD * 4);
    float* S   = (float*)wsAlloc(NBB * 4);
    bf16*  SM  = (bf16*) wsAlloc(NBB * 2);

    zero_f32_kernel<<<dim3((NB * BD + 255) / 256), 256, 0, stream>>>(DK2, NB * BD);
    zero_f32_kernel<<<dim3((NB * BD + 255) / 256), 256, 0, stream>>>(DQ2, NB * BD);

    cast_f32_bf16_kernel<<<dim3((unsigned)(NCB / 4 / 256)), 256, 0, stream>>>(X, Xb, (int)(NCB / 4));
    cast_f32_bf16_kernel<<<dim3((unsigned)(CC / 4 / 256)),  256, 0, stream>>>(Wk, Wkb, (int)(CC / 4));
    cast_f32_bf16_kernel<<<dim3((unsigned)(CC / 4 / 256)),  256, 0, stream>>>(Wq, Wqb, (int)(CC / 4));

    dim3 g(BD / 128, CD / 128, NB), blk(256);

    gemm_wmma_kernel<0><<<g, blk, 0, stream>>>(Wkb, 0, Xb, (size_t)CD * BD,
                                               (void*)Kb, Wk0, nullptr, DK2);
    gemm_wmma_kernel<0><<<g, blk, 0, stream>>>(Wqb, 0, Xb, (size_t)CD * BD,
                                               (void*)Qb, Wq0, nullptr, DQ2);
    gemm_wmma_kernel<1><<<g, blk, 0, stream>>>(Qb, (size_t)CD * BD, Kb, (size_t)CD * BD,
                                               (void*)S, DQ2, DK2, nullptr);
    softmax_col_kernel<<<dim3(BD / 256, NB), 256, 0, stream>>>(S, SM);
    gemm_wmma_kernel<2><<<g, blk, 0, stream>>>(Xb, (size_t)CD * BD, SM, (size_t)BD * BD,
                                               d_out, nullptr, nullptr, nullptr);
}
